// AttentionLayer_26096221290887
// MI455X (gfx1250) — compile-verified, matrix operand order checked by reading, not run on hardware
//
#include <hip/hip_runtime.h>
#include <hip/hip_bf16.h>
#include <math.h>
#include <stdint.h>
#include <type_traits>

// ---------------------------------------------------------------------------
// CDNA5 (gfx1250) wave32 WMMA attention:
//   stage 1: Q/K/V projections  (f32 -> f16 GEMM + bias, WMMA f32 accum)
//   stage 2: flash-attention per (b,n,h); Q/K tiles DMA'd to LDS by the
//            Tensor Data Mover (tensor_load_to_lds + s_wait_tensorcnt)
//   stage 3: output projection  (f16 -> f32 GEMM + bias)
// All WMMA operands sit in LDS with the contraction dim contiguous, so every
// fragment load is two aligned ds_load_b128s.
// ---------------------------------------------------------------------------

typedef __attribute__((ext_vector_type(16))) _Float16     v16h;
typedef __attribute__((ext_vector_type(8)))  _Float16     h8;
typedef __attribute__((ext_vector_type(4)))  _Float16     h4;
typedef __attribute__((ext_vector_type(8)))  float        v8f;
typedef __attribute__((ext_vector_type(4)))  unsigned int u32x4;
typedef __attribute__((ext_vector_type(4)))  int          i32x4;
typedef __attribute__((ext_vector_type(8)))  int          i32x8;

#define M_ROWS   65536   // B*N*L = 4*64*256
#define IN_DIM   128
#define H_DIM    512
#define N_HEADS  8
#define HD       64      // head dim
#define LQ       256     // sequence length

__device__ __forceinline__ v8f wmma_f16(v16h a, v16h b, v8f c) {
    return __builtin_amdgcn_wmma_f32_16x16x32_f16(false, a, false, b, (short)0, c,
                                                  false, false);
}

// 16x32 f16 fragment from an LDS tile stored row-major with K contiguous.
// ISA 7.12.2 mapping: elem e<8 -> K = hi*8+e ; e>=8 -> K = 16+hi*8+(e-8).
// ld must be a multiple of 8 halfs (16 B) and k_base a multiple of 32.
__device__ __forceinline__ v16h load_frag(const _Float16* lds, int row_base,
                                          int ld, int k_base, int lane) {
    const int m  = lane & 15;
    const int hi = lane >> 4;
    const _Float16* p = lds + (row_base + m) * ld + k_base + hi * 8;
    const h8 lo  = *(const h8*)(p);       // K = hi*8 .. hi*8+7
    const h8 hi8 = *(const h8*)(p + 16);  // K = 16+hi*8 .. 16+hi*8+7
    v16h f;
#pragma unroll
    for (int e = 0; e < 8; ++e) { f[e] = lo[e]; f[e + 8] = hi8[e]; }
    return f;
}

// ---------------------------------------------------------------------------
// Tensor Data Mover: DMA a 2D f16 tile (tile_h rows x 64 halfs, global row
// stride 512 halfs) into LDS.  pad_interval=32 DW (one 128 B row) +
// pad_amount=4 DW (16 B) gives an LDS row stride of 144 B = 72 halfs = LDH,
// matching the fragment loader's alignment requirements.
// D# layout per CDNA5 ISA.  Issue once per wave.
// ---------------------------------------------------------------------------
__device__ __forceinline__ void tdm_load_tile_f16(unsigned lds_byte_addr,
                                                  const _Float16* gsrc) {
    const unsigned long long ga = (unsigned long long)(uintptr_t)gsrc;
    u32x4 g0;
    g0[0] = 1u;                                   // count=1, user load D#
    g0[1] = lds_byte_addr;                        // LDS destination (bytes)
    g0[2] = (unsigned)(ga & 0xffffffffu);         // global_addr[31:0]
    g0[3] = (unsigned)((ga >> 32) & 0x01ffffffu)  // global_addr[56:32]
          | (2u << 30);                           // type = 2 (image)
    i32x8 g1;
    g1[0] = (1 << 16)      // data_size = 2 bytes
          | (1 << 20)      // pad_enable
          | (4 << 22)      // pad_interval code 4 = 32 DWORDs
          | (3 << 25);     // pad_amount  code 3 = 4 DWORDs
    g1[1] = (H_DIM << 16); // tensor_dim0[15:0]=512 (abar_addr=0)
    g1[2] = 0;             // tensor_dim0[31:16]=0 | tensor_dim1[15:0]=0
    g1[3] = 1 | (64 << 16);// tensor_dim1[31:16]=1 (dim1=65536) | tile_dim0=64
    g1[4] = 64;            // tile_dim1=64 rows | tile_dim2=0
    g1[5] = H_DIM;         // tensor_dim0_stride[31:0] = 512 elems
    g1[6] = 0;             // tensor_dim0_stride[47:32] | tensor_dim1_stride lo
    g1[7] = 0;
    const i32x4 gz = {0, 0, 0, 0};
#if defined(__clang_major__) && (__clang_major__ >= 23)
    const i32x8 gz8 = {0, 0, 0, 0, 0, 0, 0, 0};
    __builtin_amdgcn_tensor_load_to_lds(g0, g1, gz, gz, gz8, 0);
#else
    __builtin_amdgcn_tensor_load_to_lds(g0, g1, gz, gz, 0);
#endif
}

// ---------------------------------------------------------------------------
// C[M,N] = A[M,K] @ W[K,N] + bias.  Block tile 256x64, 8 waves, each wave
// owns 32 rows x 64 cols (2 m-tiles x 4 n-tiles = 8 WMMA per K-step of 32).
// Register-staged double buffering; W tiles stored transposed in LDS.
// ---------------------------------------------------------------------------
template <typename AT, typename OT>
__global__ void __launch_bounds__(256)
gemm_bias(const AT* __restrict__ A, const float* __restrict__ W,
          const float* __restrict__ bias, OT* __restrict__ C,
          int M, int K, int N) {
    constexpr bool A_F32 = std::is_same<AT, float>::value;
    constexpr int LDA = 40;                 // 32 K + pad; 80 B stride
    __shared__ _Float16 sA[2][256 * LDA];
    __shared__ _Float16 sBt[2][64 * LDA];

    const int tid  = threadIdx.x;
    const int lane = tid & 31;
    const int wave = tid >> 5;
    const int rowBase = blockIdx.x * 256;
    const int colBase = blockIdx.y * 64;
    const int nSteps  = K >> 5;

    v8f acc[2][4];
#pragma unroll
    for (int mi = 0; mi < 2; ++mi)
#pragma unroll
        for (int t = 0; t < 4; ++t) acc[mi][t] = v8f{};

    float4 ra[8];
    h8     rah[4];
    float4 rb[2];

    auto loadRegs = [&](int ks) {
        if constexpr (A_F32) {
#pragma unroll
            for (int j = 0; j < 8; ++j) {
                const int v = tid + j * 256;
                const int r = v >> 3, c = (v & 7) * 4;
                ra[j] = *(const float4*)&A[(size_t)(rowBase + r) * K + ks + c];
            }
        } else {
#pragma unroll
            for (int j = 0; j < 4; ++j) {
                const int v = tid + j * 256;
                const int r = v >> 2, c = (v & 3) * 8;
                rah[j] = *(const h8*)&A[(size_t)(rowBase + r) * K + ks + c];
            }
        }
#pragma unroll
        for (int j = 0; j < 2; ++j) {
            const int v = tid + j * 256;
            const int kk = v >> 4, n = (v & 15) * 4;
            rb[j] = *(const float4*)&W[(size_t)(ks + kk) * N + colBase + n];
        }
    };
    auto storeSmem = [&](int buf) {
        if constexpr (A_F32) {
#pragma unroll
            for (int j = 0; j < 8; ++j) {
                const int v = tid + j * 256;
                const int r = v >> 3, c = (v & 7) * 4;
                h4 t = {(_Float16)ra[j].x, (_Float16)ra[j].y,
                        (_Float16)ra[j].z, (_Float16)ra[j].w};
                *(h4*)&sA[buf][r * LDA + c] = t;
            }
        } else {
#pragma unroll
            for (int j = 0; j < 4; ++j) {
                const int v = tid + j * 256;
                const int r = v >> 2, c = (v & 3) * 8;
                *(h8*)&sA[buf][r * LDA + c] = rah[j];
            }
        }
#pragma unroll
        for (int j = 0; j < 2; ++j) {
            const int v = tid + j * 256;
            const int kk = v >> 4, n = (v & 15) * 4;
            sBt[buf][(n + 0) * LDA + kk] = (_Float16)rb[j].x;
            sBt[buf][(n + 1) * LDA + kk] = (_Float16)rb[j].y;
            sBt[buf][(n + 2) * LDA + kk] = (_Float16)rb[j].z;
            sBt[buf][(n + 3) * LDA + kk] = (_Float16)rb[j].w;
        }
    };

    loadRegs(0);
    storeSmem(0);
    __syncthreads();

    for (int s = 0; s < nSteps; ++s) {
        const int cur = s & 1;
        if (s + 1 < nSteps) loadRegs((s + 1) * 32);
        if (s + 2 < nSteps)  // L2 prefetch two tiles ahead (global_prefetch_b8)
            __builtin_prefetch(&A[(size_t)(rowBase + tid) * K + (s + 2) * 32], 0, 3);

        const v16h a0 = load_frag(sA[cur], wave * 32,      LDA, 0, lane);
        const v16h a1 = load_frag(sA[cur], wave * 32 + 16, LDA, 0, lane);
#pragma unroll
        for (int t = 0; t < 4; ++t) {
            const v16h bf = load_frag(sBt[cur], t * 16, LDA, 0, lane);
            acc[0][t] = wmma_f16(a0, bf, acc[0][t]);
            acc[1][t] = wmma_f16(a1, bf, acc[1][t]);
        }
        if (s + 1 < nSteps) storeSmem(1 - cur);
        __syncthreads();
    }

    const int hi = lane >> 4;
#pragma unroll
    for (int mi = 0; mi < 2; ++mi)
#pragma unroll
        for (int t = 0; t < 4; ++t) {
            const int col = colBase + t * 16 + (lane & 15);
            const float bv = bias[col];
#pragma unroll
            for (int r = 0; r < 8; ++r) {
                const int row = rowBase + wave * 32 + mi * 16 + r + 8 * hi;
                C[(size_t)row * N + col] = (OT)(acc[mi][t][r] + bv);
            }
        }
}

// ---------------------------------------------------------------------------
// Flash attention. grid = (Lq/64, B*N*H).  Q/K tiles arrive via the TDM with
// hardware row padding to LDH=72 halfs; V is staged transposed by the wave.
// Wave w owns tiles (mt = w>>1, nt in {2*(w&1), +1}) of both S and O.
// ---------------------------------------------------------------------------
__global__ void __launch_bounds__(256)
flash_attn(const _Float16* __restrict__ qw, const _Float16* __restrict__ kw,
           const _Float16* __restrict__ vw, _Float16* __restrict__ ow) {
    constexpr int LDH  = 72;  // 144 B stride (matches TDM pad: 128 B + 16 B)
    constexpr int LDSS = 68;
    __shared__ _Float16 sQ[64 * LDH];   // [q_row][k]      (TDM)
    __shared__ _Float16 sK[64 * LDH];   // [k_row][k]      (TDM)
    __shared__ _Float16 sVt[64 * LDH];  // [d][k_row]      (manual transpose)
    __shared__ _Float16 sP[64 * LDH];   // [q_row][k_row]
    __shared__ float    sS[64 * LDSS];
    __shared__ float    m_st[64], l_st[64], alpha[64];

    const int tid  = threadIdx.x;
    const int lane = tid & 31;
    const int wave = tid >> 5;
    const int bnh  = blockIdx.y;
    const int bn   = bnh >> 3;
    const int h    = bnh & 7;
    const size_t rowBase = (size_t)bn * LQ;
    const int colH = h * HD;
    const int q0   = blockIdx.x * 64;

    // Q tile: one TDM descriptor issued by wave 0 (TDM ignores EXEC; the
    // uniform branch keeps it to a single issue for the workgroup).
    if (wave == 0) {
        tdm_load_tile_f16((unsigned)(uintptr_t)&sQ[0],
                          &qw[(rowBase + q0) * H_DIM + colH]);
    }
    if (tid < 64) { m_st[tid] = -INFINITY; l_st[tid] = 0.f; }

    const int mt  = wave >> 1;
    const int nt0 = (wave & 1) * 2;
    const int hi  = lane >> 4;
    v8f accO[2] = {v8f{}, v8f{}};

    for (int kt = 0; kt < 4; ++kt) {
        __syncthreads();  // prior iteration done reading sK/sVt/sP
        if (wave == 0) {
            tdm_load_tile_f16((unsigned)(uintptr_t)&sK[0],
                              &kw[(rowBase + kt * 64) * H_DIM + colH]);
        }
        // V tile staged transposed (d-major) so P@V B-frags stay b128 loads
#pragma unroll
        for (int j = 0; j < 2; ++j) {
            const int v = tid + j * 256;
            const int r = v >> 3, c = (v & 7) * 8;
            const h8 vv =
                *(const h8*)&vw[(rowBase + kt * 64 + r) * H_DIM + colH + c];
#pragma unroll
            for (int u = 0; u < 8; ++u) sVt[(c + u) * LDH + r] = vv[u];
        }
        __builtin_amdgcn_s_wait_tensorcnt(0);  // Q (first iter) + K tile landed
        __syncthreads();

        // S = (Q @ K^T) / sqrt(hd); K^T fragment == row-major load of sK
#pragma unroll
        for (int j = 0; j < 2; ++j) {
            const int nt = nt0 + j;
            v8f s = {};
#pragma unroll
            for (int ks = 0; ks < 2; ++ks) {
                const v16h a = load_frag(sQ, mt * 16, LDH, ks * 32, lane);
                const v16h b = load_frag(sK, nt * 16, LDH, ks * 32, lane);
                s = wmma_f16(a, b, s);
            }
            const int col = nt * 16 + (lane & 15);
#pragma unroll
            for (int r = 0; r < 8; ++r)
                sS[(mt * 16 + r + 8 * hi) * LDSS + col] = s[r] * 0.125f;
        }
        __syncthreads();

        // running softmax: one lane per query row
        if (tid < 64) {
            const float mo = m_st[tid];
            float mx = mo;
            for (int c = 0; c < 64; ++c) mx = fmaxf(mx, sS[tid * LDSS + c]);
            const float al = __expf(mo - mx);
            float sum = 0.f;
            for (int c = 0; c < 64; ++c) {
                const float p = __expf(sS[tid * LDSS + c] - mx);
                sP[tid * LDH + c] = (_Float16)p;
                sum += p;
            }
            m_st[tid]  = mx;
            l_st[tid]  = l_st[tid] * al + sum;
            alpha[tid] = al;
        }
        __syncthreads();

        // rescale running O, accumulate P @ V
#pragma unroll
        for (int j = 0; j < 2; ++j) {
#pragma unroll
            for (int r = 0; r < 8; ++r)
                accO[j][r] *= alpha[mt * 16 + r + 8 * hi];
            const int nt = nt0 + j;
#pragma unroll
            for (int ks = 0; ks < 2; ++ks) {
                const v16h a = load_frag(sP,  mt * 16, LDH, ks * 32, lane);
                const v16h b = load_frag(sVt, nt * 16, LDH, ks * 32, lane);
                accO[j] = wmma_f16(a, b, accO[j]);
            }
        }
    }
    __syncthreads();

    // normalize by row sums, store f16 head output
#pragma unroll
    for (int j = 0; j < 2; ++j) {
        const int col = colH + (nt0 + j) * 16 + (lane & 15);
#pragma unroll
        for (int r = 0; r < 8; ++r) {
            const int rl = mt * 16 + r + 8 * hi;
            const float inv = 1.0f / l_st[rl];
            ow[(rowBase + q0 + rl) * H_DIM + col] = (_Float16)(accO[j][r] * inv);
        }
    }
}

// ---------------------------------------------------------------------------
extern "C" void kernel_launch(void* const* d_in, const int* in_sizes, int n_in,
                              void* d_out, int out_size, void* d_ws, size_t ws_size,
                              hipStream_t stream) {
    (void)in_sizes; (void)n_in; (void)out_size; (void)ws_size;
    const float* query = (const float*)d_in[0];
    const float* key   = (const float*)d_in[1];
    const float* value = (const float*)d_in[2];
    const float* Wq    = (const float*)d_in[3];
    const float* bq    = (const float*)d_in[4];
    const float* Wk    = (const float*)d_in[5];
    const float* bk    = (const float*)d_in[6];
    const float* Wv    = (const float*)d_in[7];
    const float* bv    = (const float*)d_in[8];
    const float* Wo    = (const float*)d_in[9];
    const float* bo    = (const float*)d_in[10];
    float* out = (float*)d_out;

    const size_t ten = (size_t)M_ROWS * H_DIM * sizeof(_Float16);
    char* ws = (char*)d_ws;
    _Float16* q_ws = (_Float16*)(ws);
    _Float16* k_ws = (_Float16*)(ws + ten);
    _Float16* v_ws = (_Float16*)(ws + 2 * ten);
    _Float16* a_ws = (_Float16*)(ws + 3 * ten);

    const dim3 blk(256);
    const dim3 gProj(M_ROWS / 256, H_DIM / 64);   // 256 x 8

    gemm_bias<float, _Float16><<<gProj, blk, 0, stream>>>(query, Wq, bq, q_ws,
                                                          M_ROWS, IN_DIM, H_DIM);
    gemm_bias<float, _Float16><<<gProj, blk, 0, stream>>>(key,   Wk, bk, k_ws,
                                                          M_ROWS, IN_DIM, H_DIM);
    gemm_bias<float, _Float16><<<gProj, blk, 0, stream>>>(value, Wv, bv, v_ws,
                                                          M_ROWS, IN_DIM, H_DIM);

    flash_attn<<<dim3(LQ / 64, 4 * 64 * N_HEADS), blk, 0, stream>>>(q_ws, k_ws,
                                                                    v_ws, a_ws);

    gemm_bias<_Float16, float><<<gProj, blk, 0, stream>>>(a_ws, Wo, bo, out,
                                                          M_ROWS, H_DIM, H_DIM);
}
